// LongContextMultiHeadAttention_27075473834352
// MI455X (gfx1250) — matrix operand order repaired and checked by reading, compiled, not run on hardware
//
#include <hip/hip_runtime.h>

#define D_MODEL   2048
#define NUM_HEADS 16
#define HEAD_DIM  128
#define SEQ       2048
#define BATCH     2
#define M_TOTAL   (BATCH * SEQ)   // 4096

typedef __bf16 bf16_t;
typedef __attribute__((ext_vector_type(16))) __bf16 v16bf;
typedef __attribute__((ext_vector_type(8)))  float  v8f;

union Frag16 { v16bf v; uint4 u[2]; };

// ---------------------------------------------------------------------------
// CDNA5 async global->LDS copy (GLOBAL_LOAD_ASYNC_TO_LDS_B128, ASYNCcnt).
// ---------------------------------------------------------------------------
#if defined(__AMDGCN__) && \
    __has_builtin(__builtin_amdgcn_global_load_async_to_lds_b128) && \
    __has_builtin(__builtin_amdgcn_s_wait_asynccnt)
#define USE_ASYNC_LDS 1
#else
#define USE_ASYNC_LDS 0
#endif

typedef int b128_t __attribute__((vector_size(16)));

__device__ inline void async_cp16(void* lds, const void* g) {
#if USE_ASYNC_LDS
    __builtin_amdgcn_global_load_async_to_lds_b128(
        (__attribute__((address_space(1))) b128_t*)g,
        (__attribute__((address_space(3))) b128_t*)lds, 0, 0);
#else
    *(uint4*)lds = *(const uint4*)g;
#endif
}

__device__ inline void wait_stage_barrier() {
#if USE_ASYNC_LDS
    __builtin_amdgcn_s_wait_asynccnt(0);
#endif
    __syncthreads();
}

// ---------------------------------------------------------------------------
// Elementwise fp32 -> bf16 (one-time conversion so GEMM staging is pure copy).
// ---------------------------------------------------------------------------
__global__ __launch_bounds__(256) void cvt_kernel(
    const float* __restrict__ s, bf16_t* __restrict__ d)
{
    const size_t i = ((size_t)blockIdx.x * 256 + threadIdx.x) * 8;
    const float4 f0 = *(const float4*)(s + i);
    const float4 f1 = *(const float4*)(s + i + 4);
    union { bf16_t b[8]; uint4 u; } r;
    r.b[0] = (bf16_t)f0.x; r.b[1] = (bf16_t)f0.y;
    r.b[2] = (bf16_t)f0.z; r.b[3] = (bf16_t)f0.w;
    r.b[4] = (bf16_t)f1.x; r.b[5] = (bf16_t)f1.y;
    r.b[6] = (bf16_t)f1.z; r.b[7] = (bf16_t)f1.w;
    *(uint4*)(d + i) = r.u;
}

// ---------------------------------------------------------------------------
// GEMM: out[m,n] = (sum_k A[m,k]*W[n,k] + bias[n]) * ascale
// MODE 0: scatter bf16 into [B,H,S,HEAD_DIM]; MODE 1: fp32 linear out.
// Block 256 thr (8 waves), tile 128x128; wave tile 32x64 = 8 WMMA/K-step.
// Double-buffered LDS, async staging, one barrier per K-step.
// ---------------------------------------------------------------------------
template <int MODE>
__global__ __launch_bounds__(256) void gemm_bf16_kernel(
    const bf16_t* __restrict__ A,
    const bf16_t* __restrict__ W,
    const float*  __restrict__ bias,
    void* __restrict__ outp,
    float ascale)
{
    __shared__ bf16_t At[2][128][40];
    __shared__ bf16_t Bt[2][128][40];

    const int t    = threadIdx.x;
    const int lane = t & 31;
    const int wave = t >> 5;
    const int wm   = wave & 3;
    const int wn   = wave >> 2;
    const int m0   = blockIdx.y * 128;
    const int n0   = blockIdx.x * 128;

    const int srow = t >> 1;          // 0..127
    const int scol = (t & 1) * 16;    // 0 or 16
    const int lr   = lane & 15;
    const int kbA  = (lane < 16) ? 0 : 8;
    const int kbB  = (lane < 16) ? 0 : 8;

    auto stage = [&](int buf, int k0) {
        const bf16_t* as = A + (size_t)(m0 + srow) * D_MODEL + k0 + scol;
        const bf16_t* ws = W + (size_t)(n0 + srow) * D_MODEL + k0 + scol;
        async_cp16(&At[buf][srow][scol],     as);
        async_cp16(&At[buf][srow][scol + 8], as + 8);
        async_cp16(&Bt[buf][srow][scol],     ws);
        async_cp16(&Bt[buf][srow][scol + 8], ws + 8);
    };

    v8f acc[2][4] = {};
    stage(0, 0);

    int buf = 0;
    for (int k0 = 0; k0 < D_MODEL; k0 += 32, buf ^= 1) {
        wait_stage_barrier();
        if (k0 + 32 < D_MODEL) stage(buf ^ 1, k0 + 32);

        Frag16 a[2];
#pragma unroll
        for (int i = 0; i < 2; ++i) {
            const uint4* ap = (const uint4*)&At[buf][wm * 32 + i * 16 + lr][0];
            a[i].u[0] = ap[kbA >> 3];
            a[i].u[1] = ap[(kbA + 16) >> 3];
        }
#pragma unroll
        for (int j = 0; j < 4; ++j) {
            Frag16 b;
            const uint4* bp = (const uint4*)&Bt[buf][wn * 64 + j * 16 + lr][0];
            b.u[0] = bp[kbB >> 3];
            b.u[1] = bp[(kbB >> 3) + 2];
#pragma unroll
            for (int i = 0; i < 2; ++i)
                acc[i][j] = __builtin_amdgcn_wmma_f32_16x16x32_bf16(
                    false, a[i].v, false, b.v, (short)0, acc[i][j], false, false);
        }
    }

    const int rowoff = (lane < 16) ? 0 : 8;
#pragma unroll
    for (int i = 0; i < 2; ++i) {
#pragma unroll
        for (int j = 0; j < 4; ++j) {
            const int ng   = n0 + wn * 64 + j * 16 + lr;
            const float bv = bias[ng];
#pragma unroll
            for (int r = 0; r < 8; ++r) {
                const int m     = m0 + wm * 32 + i * 16 + rowoff + r;
                const float val = (acc[i][j][r] + bv) * ascale;
                if (MODE == 0) {
                    const int hh = ng >> 7, dd = ng & 127;
                    const int bidx = m >> 11, sg = m & (SEQ - 1);
                    ((bf16_t*)outp)[(((size_t)bidx * NUM_HEADS + hh) * SEQ + sg) * HEAD_DIM + dd] =
                        (bf16_t)val;
                } else {
                    ((float*)outp)[(size_t)m * D_MODEL + ng] = val;
                }
            }
        }
    }
}

// ---------------------------------------------------------------------------
// Flash attention, one (b, h, 128-row q tile) per block. 8 waves x 16 q rows,
// Q fragments resident in VGPRs (Q pre-scaled by 1/sqrt(d)).
// KV tiles of 64 rows, double-buffered: K async straight copy, V transposed
// with packed dword stores. One barrier per tile; 32 WMMAs/wave/tile.
// ---------------------------------------------------------------------------
__global__ __launch_bounds__(256) void attn_kernel(
    const bf16_t* __restrict__ Qh,   // [B,H,S,HEAD_DIM]
    const bf16_t* __restrict__ Kh,
    const bf16_t* __restrict__ Vh,
    bf16_t* __restrict__ Oh)         // [B,S,D_MODEL]
{
    __shared__ bf16_t Kbt[2][64][136];   // K rows (BT layout), double-buffered
    __shared__ bf16_t Vt[2][128][72];    // V transposed: Vt[d][n], double-buffered
    __shared__ bf16_t Ps[8][16][72];     // per-wave P scratch (C->A relayout)

    const int t    = threadIdx.x;
    const int lane = t & 31;
    const int wave = t >> 5;
    const int h    = blockIdx.y;
    const int b    = blockIdx.z;
    const size_t headbase = ((size_t)b * NUM_HEADS + h) * (size_t)SEQ * HEAD_DIM;

    const int lr  = lane & 15;
    const int kbA = (lane < 16) ? 0 : 8;
    const int kb2 = (lane < 16) ? 0 : 8;
    const int q0  = blockIdx.x * 128 + wave * 16;

    Frag16 qa[4];
    {
        const uint4* qrow = (const uint4*)(Qh + headbase + (size_t)(q0 + lr) * HEAD_DIM);
#pragma unroll
        for (int tt = 0; tt < 4; ++tt) {
            qa[tt].u[0] = qrow[(tt * 32 + kbA) >> 3];
            qa[tt].u[1] = qrow[(tt * 32 + kbA + 16) >> 3];
        }
    }

    float mrun[8], lrun[8], alpha[8];
#pragma unroll
    for (int r = 0; r < 8; ++r) { mrun[r] = -1e30f; lrun[r] = 0.0f; }
    v8f o[8] = {};

    const int rn2 = (t >> 3) * 2;        // 0,2,...,62  (pair of kv rows)
    const int c16 = (t & 7) * 16;        // 0..112      (16-wide d slice)

    auto stage_kv = [&](int bufi, int kv0) {
        // K rows: straight async copies
        const bf16_t* ks0 = Kh + headbase + (size_t)(kv0 + rn2) * HEAD_DIM + c16;
        const bf16_t* ks1 = ks0 + HEAD_DIM;
        async_cp16(&Kbt[bufi][rn2][c16],         ks0);
        async_cp16(&Kbt[bufi][rn2][c16 + 8],     ks0 + 8);
        async_cp16(&Kbt[bufi][rn2 + 1][c16],     ks1);
        async_cp16(&Kbt[bufi][rn2 + 1][c16 + 8], ks1 + 8);
        // V rows: transpose with paired dword stores
        const bf16_t* vs0 = Vh + headbase + (size_t)(kv0 + rn2) * HEAD_DIM + c16;
        const bf16_t* vs1 = vs0 + HEAD_DIM;
        union { uint4 q[2]; unsigned short s[16]; } r0, r1;
        r0.q[0] = *(const uint4*)vs0; r0.q[1] = *(const uint4*)(vs0 + 8);
        r1.q[0] = *(const uint4*)vs1; r1.q[1] = *(const uint4*)(vs1 + 8);
#pragma unroll
        for (int i = 0; i < 16; ++i) {
            const unsigned int pk = (unsigned int)r0.s[i] | ((unsigned int)r1.s[i] << 16);
            *(unsigned int*)&Vt[bufi][c16 + i][rn2] = pk;
        }
    };

    stage_kv(0, 0);
    int buf = 0;
    for (int kv0 = 0; kv0 < SEQ; kv0 += 64, buf ^= 1) {
        wait_stage_barrier();                      // tile `buf` ready everywhere
        if (kv0 + 64 < SEQ) stage_kv(buf ^ 1, kv0 + 64);

        // ---- scores: S(16x64) = Q(16x128) x K^T ----
        v8f sf[4] = {};
#pragma unroll
        for (int tt = 0; tt < 4; ++tt) {
#pragma unroll
            for (int f = 0; f < 4; ++f) {
                Frag16 bk;
                const uint4* kp = (const uint4*)&Kbt[buf][f * 16 + lr][0];
                const int idx = (tt * 32 + kb2) >> 3;
                bk.u[0] = kp[idx];
                bk.u[1] = kp[idx + 2];
                sf[f] = __builtin_amdgcn_wmma_f32_16x16x32_bf16(
                    false, qa[tt].v, false, bk.v, (short)0, sf[f], false, false);
            }
        }

        // ---- online softmax (rows live in 16-lane groups) ----
        const int rowoff = (lane < 16) ? 0 : 8;
#pragma unroll
        for (int r = 0; r < 8; ++r) {
            float mx = fmaxf(fmaxf(sf[0][r], sf[1][r]), fmaxf(sf[2][r], sf[3][r]));
#pragma unroll
            for (int msk = 1; msk < 16; msk <<= 1)
                mx = fmaxf(mx, __shfl_xor(mx, msk, 32));
            const float mnew = fmaxf(mrun[r], mx);
            const float al   = __expf(mrun[r] - mnew);
            mrun[r]  = mnew;
            alpha[r] = al;
            float ls = 0.0f;
#pragma unroll
            for (int f = 0; f < 4; ++f) {
                const float e = __expf(sf[f][r] - mnew);
                ls += e;
                Ps[wave][rowoff + r][f * 16 + lr] = (bf16_t)e;
            }
#pragma unroll
            for (int msk = 1; msk < 16; msk <<= 1)
                ls += __shfl_xor(ls, msk, 32);
            lrun[r] = lrun[r] * al + ls;
        }
#pragma unroll
        for (int j = 0; j < 8; ++j)
#pragma unroll
            for (int r = 0; r < 8; ++r) o[j][r] = o[j][r] * alpha[r];

        // ---- P: C-layout -> two A-frags (same-wave LDS, in-order) ----
        Frag16 pa0, pa1;
        {
            const uint4* pp = (const uint4*)&Ps[wave][lr][0];
            pa0.u[0] = pp[kbA >> 3];
            pa0.u[1] = pp[(kbA + 16) >> 3];
            pa1.u[0] = pp[(32 + kbA) >> 3];
            pa1.u[1] = pp[(48 + kbA) >> 3];
        }

        // ---- O(16x128) += P(16x64) x V(64x128) ----
#pragma unroll
        for (int j = 0; j < 8; ++j) {
            const uint4* vp = (const uint4*)&Vt[buf][j * 16 + lr][0];
            Frag16 bv0, bv1;
            bv0.u[0] = vp[kb2 >> 3];
            bv0.u[1] = vp[(kb2 >> 3) + 2];
            bv1.u[0] = vp[(32 + kb2) >> 3];
            bv1.u[1] = vp[((32 + kb2) >> 3) + 2];
            o[j] = __builtin_amdgcn_wmma_f32_16x16x32_bf16(
                false, pa0.v, false, bv0.v, (short)0, o[j], false, false);
            o[j] = __builtin_amdgcn_wmma_f32_16x16x32_bf16(
                false, pa1.v, false, bv1.v, (short)0, o[j], false, false);
        }
        // no trailing barrier: next tile writes go to the other buffer
    }

    // ---- normalize + store bf16 to [B,S,D] layout ----
    {
        const int rowoff = (lane < 16) ? 0 : 8;
        float inv[8];
#pragma unroll
        for (int r = 0; r < 8; ++r) inv[r] = 1.0f / lrun[r];
#pragma unroll
        for (int j = 0; j < 8; ++j)
#pragma unroll
            for (int r = 0; r < 8; ++r) {
                const int sg = q0 + rowoff + r;
                Oh[((size_t)b * SEQ + sg) * D_MODEL + h * HEAD_DIM + j * 16 + lr] =
                    (bf16_t)(o[j][r] * inv[r]);
            }
    }
}

// ---------------------------------------------------------------------------
extern "C" void kernel_launch(void* const* d_in, const int* in_sizes, int n_in,
                              void* d_out, int out_size, void* d_ws, size_t ws_size,
                              hipStream_t stream) {
    (void)in_sizes; (void)n_in; (void)out_size; (void)ws_size;
    const float* q  = (const float*)d_in[0];
    const float* k  = (const float*)d_in[1];
    const float* v  = (const float*)d_in[2];
    const float* Wq = (const float*)d_in[3];
    const float* bq = (const float*)d_in[4];
    const float* Wk = (const float*)d_in[5];
    const float* bk = (const float*)d_in[6];
    const float* Wv = (const float*)d_in[7];
    const float* bv = (const float*)d_in[8];
    const float* Wo = (const float*)d_in[9];
    const float* bo = (const float*)d_in[10];
    float* out = (float*)d_out;

    const size_t nx = (size_t)M_TOTAL * D_MODEL;
    const size_t nw = (size_t)D_MODEL * D_MODEL;

    bf16_t* p   = (bf16_t*)d_ws;
    bf16_t* xqb = p; p += nx;
    bf16_t* xkb = p; p += nx;
    bf16_t* xvb = p; p += nx;
    bf16_t* Wqb = p; p += nw;
    bf16_t* Wkb = p; p += nw;
    bf16_t* Wvb = p; p += nw;
    bf16_t* Wob = p; p += nw;
    bf16_t* Qh  = p; p += nx;
    bf16_t* Kh  = p; p += nx;
    bf16_t* Vh  = p; p += nx;
    bf16_t* Ah  = p; p += nx;

    dim3 blk(256);
    cvt_kernel<<<dim3(nx / 2048), blk, 0, stream>>>(q,  xqb);
    cvt_kernel<<<dim3(nx / 2048), blk, 0, stream>>>(k,  xkb);
    cvt_kernel<<<dim3(nx / 2048), blk, 0, stream>>>(v,  xvb);
    cvt_kernel<<<dim3(nw / 2048), blk, 0, stream>>>(Wq, Wqb);
    cvt_kernel<<<dim3(nw / 2048), blk, 0, stream>>>(Wk, Wkb);
    cvt_kernel<<<dim3(nw / 2048), blk, 0, stream>>>(Wv, Wvb);
    cvt_kernel<<<dim3(nw / 2048), blk, 0, stream>>>(Wo, Wob);

    const float qscale = 0.08838834764831845f;  // 1/sqrt(HEAD_DIM)
    dim3 gproj(D_MODEL / 128, M_TOTAL / 128);
    gemm_bf16_kernel<0><<<gproj, blk, 0, stream>>>(xqb, Wqb, bq, (void*)Qh, qscale);
    gemm_bf16_kernel<0><<<gproj, blk, 0, stream>>>(xkb, Wkb, bk, (void*)Kh, 1.0f);
    gemm_bf16_kernel<0><<<gproj, blk, 0, stream>>>(xvb, Wvb, bv, (void*)Vh, 1.0f);

    dim3 gattn(SEQ / 128, NUM_HEADS, BATCH);
    attn_kernel<<<gattn, blk, 0, stream>>>(Qh, Kh, Vh, Ah);

    gemm_bf16_kernel<1><<<gproj, blk, 0, stream>>>(Ah, Wob, bo, (void*)out, 1.0f);
}
